// GCN_7928509628445
// MI455X (gfx1250) — matrix operand order repaired and checked by reading, not compiled, not run on hardware
//
#include <hip/hip_runtime.h>
#include <hip/hip_bf16.h>

// ---------------------------------------------------------------------------
// GCN forward on MI455X (gfx1250, wave32, WMMA + async-to-LDS staging).
//   An = D^-1/2 (A+I) D^-1/2 is never materialized:
//     T' = rowscale(X@W, dinv)   (bf16, stored transposed -> B-fragment layout)
//     Z  = A @ T' + T'           (big bf16 WMMA GEMM; B slab double-buffered in
//                                 LDS via global_load_async_to_lds_b128)
//     X' = relu(rowscale(Z,dinv) + b)
//   If ws_size allows, adjacency is converted once to a K-padded bf16 copy
//   (fused into the degree pass): 400MB/layer -> 200MB/layer of HBM traffic
//   and no per-k-step cvt VALU in the dominant kernel.
// ---------------------------------------------------------------------------

typedef __attribute__((ext_vector_type(16))) __bf16 v16bf;
typedef __attribute__((ext_vector_type(8)))  __bf16 v8bf;
typedef __attribute__((ext_vector_type(8)))  float  v8f;

#define NNODES 10000
#define KP     10016    // padded K (mult of 32); cols 10000..10015 zeroed
#define HDIM   256
#define DIN    1024

// two async 16B global->LDS copies (32B per thread, ASYNCcnt-tracked)
#define ASYNC_COPY_32B(ldsoff_u32, gptr)                                \
  asm volatile("global_load_async_to_lds_b128 %0, %1, off\n\t"          \
               "global_load_async_to_lds_b128 %0, %1, off offset:16"    \
               :: "v"(ldsoff_u32), "v"(gptr) : "memory")

#define WAIT_ASYNC0() asm volatile("s_wait_asynccnt 0x0" ::: "memory")

#if __has_builtin(__builtin_amdgcn_sched_barrier)
#define SCHED_FENCE() __builtin_amdgcn_sched_barrier(0)  // loads stay above
#else
#define SCHED_FENCE()
#endif

// ---------------- degree / dinv (+ optional fused adj -> bf16 copy) --------
__global__ __launch_bounds__(256) void deg_cvt_kernel(
    const float* __restrict__ adj, float* __restrict__ dinv,
    __bf16* __restrict__ adjb /* nullptr -> skip convert */) {
  __shared__ float red[256];
  const int row = blockIdx.x;
  const float* r = adj + (size_t)row * NNODES;
  __bf16* ob = adjb ? adjb + (size_t)row * KP : (__bf16*)0;
  float s = 0.f;
  for (int c = threadIdx.x; c < KP / 8; c += 256) {   // 8 floats per chunk
    const int k = c * 8;
    if (k < NNODES) {
      v8f v = *(const v8f*)(r + k);
#pragma unroll
      for (int i = 0; i < 8; ++i) s += v[i];
      if (ob) {
        v8bf o;
#pragma unroll
        for (int i = 0; i < 8; ++i) o[i] = (__bf16)v[i];
        *(v8bf*)(ob + k) = o;
      }
    } else if (ob) {                                   // zero K padding
      v8bf z;
#pragma unroll
      for (int i = 0; i < 8; ++i) z[i] = (__bf16)0.0f;
      *(v8bf*)(ob + k) = z;
    }
  }
  red[threadIdx.x] = s;
  __syncthreads();
  for (int off = 128; off > 0; off >>= 1) {
    if (threadIdx.x < off) red[threadIdx.x] += red[threadIdx.x + off];
    __syncthreads();
  }
  if (threadIdx.x == 0) dinv[row] = rsqrtf(red[0] + 1.0f);  // +1 self loop
}

// ---------------- W -> Wt (bf16, transposed: [N][K]) -----------------------
__global__ __launch_bounds__(256) void twt_kernel(const float* __restrict__ W,
                                                  __bf16* __restrict__ Wt,
                                                  int K, int Nn) {
  int idx = blockIdx.x * 256 + threadIdx.x;
  if (idx >= K * Nn) return;
  int k = idx / Nn, n = idx - k * Nn;
  Wt[(size_t)n * K + k] = (__bf16)W[idx];
}

// ---------------- per-call init: Bt K-padding + pooling accumulators -------
__global__ __launch_bounds__(256) void init_kernel(__bf16* __restrict__ Bt,
                                                   unsigned* __restrict__ gmax,
                                                   float* __restrict__ gsum) {
  int idx = blockIdx.x * 256 + threadIdx.x;
  if (idx < 256 * 16) {
    int n = idx >> 4, j = idx & 15;
    Bt[(size_t)n * KP + NNODES + j] = (__bf16)0.0f;
  }
  if (idx < HDIM) { gmax[idx] = 0u; gsum[idx] = 0.f; }
}

// ---------------- GEMM1: Bt[n][m] = bf16(dinv[m] * (X @ W)[m][n]) ----------
__global__ __launch_bounds__(256) void gemm_xw_kernel(
    const float* __restrict__ X, const __bf16* __restrict__ Wt,
    const float* __restrict__ dinv, __bf16* __restrict__ Bt, int K) {
  const int lane = threadIdx.x & 31;
  const int wid  = threadIdx.x >> 5;
  const int ln   = lane & 15;
  const int h    = lane >> 4;
  const int mw   = blockIdx.x * 128 + wid * 16;   // wave row base (uniform)
  const int nB   = blockIdx.y * 128;
  if (mw >= NNODES) return;                        // wave-uniform, no barriers

  v8f acc[8];
#pragma unroll
  for (int j = 0; j < 8; ++j) acc[j] = (v8f)0.0f;

  const float* arow = X + (size_t)(mw + ln) * K;
  v8f c0 = *(const v8f*)(arow + 16 * h);
  v8f c1 = *(const v8f*)(arow + 8 + 16 * h);

  for (int k0 = 0; k0 < K; k0 += 32) {
    v16bf a;
#pragma unroll
    for (int i = 0; i < 8; ++i) { a[i] = (__bf16)c0[i]; a[8 + i] = (__bf16)c1[i]; }
    v16bf b[8];
#pragma unroll
    for (int j = 0; j < 8; ++j)
      b[j] = *(const v16bf*)(Wt + (size_t)(nB + j * 16 + ln) * K + k0 + 16 * h);
    if (k0 + 32 < K) {                 // prefetch next A chunks (issue last)
      c0 = *(const v8f*)(arow + k0 + 32 + 16 * h);
      c1 = *(const v8f*)(arow + k0 + 40 + 16 * h);
    }
    SCHED_FENCE();                     // keep all loads above the WMMA chain
#pragma unroll
    for (int j = 0; j < 8; ++j)
      acc[j] = __builtin_amdgcn_wmma_f32_16x16x32_bf16(
          false, a, false, b[j], (short)0, acc[j], false, false);
  }

  // epilogue: row-scale by dinv, write transposed bf16 (16B per lane per tile)
  v8f dv = *(const v8f*)(dinv + mw + 8 * h);
#pragma unroll
  for (int j = 0; j < 8; ++j) {
    int n = nB + j * 16 + ln;
    v8bf o;
#pragma unroll
    for (int r = 0; r < 8; ++r) o[r] = (__bf16)(dv[r] * acc[j][r]);
    *(v8bf*)(Bt + (size_t)n * KP + mw + 8 * h) = o;
  }
}

// ---------------- GEMM2: X' = relu(dinv*(A@T' + T') + b) -------------------
// ABF16=true : A is the K-padded bf16 adjacency copy (no cvt, no masking)
// ABF16=false: A is the original f32 adjacency (inline cvt + tail mask)
template <bool ABF16>
__global__ __launch_bounds__(256) void gemm_adj_kernel(
    const void* __restrict__ Asrc, const __bf16* __restrict__ Bt,
    const float* __restrict__ dinv, const float* __restrict__ bias,
    float* __restrict__ Xo) {
  constexpr int LSTR = 40;                    // bf16 per LDS row: 32 data + 8 pad
  __shared__ __bf16 lbuf[2][128 * LSTR];      // 2 x 10KB

  const int tid  = threadIdx.x;
  const int lane = tid & 31;
  const int wid  = tid >> 5;
  const int ln   = lane & 15;
  const int h    = lane >> 4;
  const int mw   = blockIdx.x * 128 + wid * 16;
  const int nB   = blockIdx.y * 128;
  const bool valid = (mw + 16) <= NNODES;     // wave-uniform
  const int  mrow  = valid ? mw : 0;          // clamp: must keep doing barriers

  // async-copy assignment: thread t moves 32B of the slab: row n=t>>1, half=t&1
  const int cn    = tid >> 1;
  const int chalf = tid & 1;
  const __bf16* gsrc = Bt + (size_t)(nB + cn) * KP + 16 * chalf;
  // flat->LDS: low 32 bits of a __shared__ flat address are the LDS byte offset
  const unsigned lds0 = (unsigned)(size_t)&lbuf[0][cn * LSTR + 16 * chalf];
  const unsigned lds1 = (unsigned)(size_t)&lbuf[1][cn * LSTR + 16 * chalf];

  v8f acc[8];
#pragma unroll
  for (int j = 0; j < 8; ++j) acc[j] = (v8f)0.0f;

  const int NK = NNODES / 32 + 1;             // 313 steps (KP/32)

  ASYNC_COPY_32B(lds0, gsrc);                 // prologue: slab 0 -> buf 0

  if constexpr (ABF16) {
    const __bf16* arow = (const __bf16*)Asrc + (size_t)(mrow + ln) * KP;
    v8bf l0 = *(const v8bf*)(arow + 8 * h);
    v8bf l1 = *(const v8bf*)(arow + 16 + 8 * h);
    for (int i = 0; i < NK; ++i) {
      const int k0 = i * 32;
      WAIT_ASYNC0();                          // own copies of buf[i&1] done
      __syncthreads();                        // everyone's done + prior reads done
      if (i + 1 < NK)
        ASYNC_COPY_32B((i & 1) ? lds0 : lds1, gsrc + k0 + 32);

      v16bf a = __builtin_shufflevector(l0, l1, 0, 1, 2, 3, 4, 5, 6, 7,
                                        8, 9, 10, 11, 12, 13, 14, 15);
      if (i + 1 < NK) {                       // pipeline next A chunks (padded!)
        l0 = *(const v8bf*)(arow + k0 + 32 + 8 * h);
        l1 = *(const v8bf*)(arow + k0 + 48 + 8 * h);
        __builtin_prefetch(arow + k0 + 544, 0, 0);
      }
      const __bf16* lb = &lbuf[i & 1][0];
      v16bf b[8];
#pragma unroll
      for (int j = 0; j < 8; ++j) {
        const __bf16* p = lb + (j * 16 + ln) * LSTR + 8 * h;
        v8bf lo = *(const v8bf*)p;
        v8bf hi = *(const v8bf*)(p + 16);
        b[j] = __builtin_shufflevector(lo, hi, 0, 1, 2, 3, 4, 5, 6, 7,
                                       8, 9, 10, 11, 12, 13, 14, 15);
      }
      SCHED_FENCE();
#pragma unroll
      for (int j = 0; j < 8; ++j)
        acc[j] = __builtin_amdgcn_wmma_f32_16x16x32_bf16(
            false, a, false, b[j], (short)0, acc[j], false, false);
    }
  } else {
    const float* arow = (const float*)Asrc + (size_t)(mrow + ln) * NNODES;
    v8f c0 = *(const v8f*)(arow + 8 * h);
    v8f c1 = *(const v8f*)(arow + 16 + 8 * h);
    for (int i = 0; i < NK; ++i) {
      const int k0 = i * 32;
      WAIT_ASYNC0();
      __syncthreads();
      if (i + 1 < NK)
        ASYNC_COPY_32B((i & 1) ? lds0 : lds1, gsrc + k0 + 32);

      v16bf a;
#pragma unroll
      for (int t2 = 0; t2 < 8; ++t2) { a[t2] = (__bf16)c0[t2]; a[8 + t2] = (__bf16)c1[t2]; }
      if (i + 1 < NK) {
        const int kn = k0 + 32;
        c0 = *(const v8f*)(arow + kn + 8 * h);
        // mask k >= NNODES: garbage * 0 would be NaN inside WMMA
        c1 = (kn + 32 <= NNODES) ? *(const v8f*)(arow + kn + 16 + 8 * h) : (v8f)0.0f;
        __builtin_prefetch(arow + kn + 256, 0, 0);
      }
      const __bf16* lb = &lbuf[i & 1][0];
      v16bf b[8];
#pragma unroll
      for (int j = 0; j < 8; ++j) {
        const __bf16* p = lb + (j * 16 + ln) * LSTR + 8 * h;
        v8bf lo = *(const v8bf*)p;
        v8bf hi = *(const v8bf*)(p + 16);
        b[j] = __builtin_shufflevector(lo, hi, 0, 1, 2, 3, 4, 5, 6, 7,
                                       8, 9, 10, 11, 12, 13, 14, 15);
      }
      SCHED_FENCE();
#pragma unroll
      for (int j = 0; j < 8; ++j)
        acc[j] = __builtin_amdgcn_wmma_f32_16x16x32_bf16(
            false, a, false, b[j], (short)0, acc[j], false, false);
    }
  }

  if (!valid) return;                         // after the last barrier: safe

  // epilogue: self-loop (+T' from Bt), dinv scale, bias, relu, f32 row-major
  v8f dv = *(const v8f*)(dinv + mw + 8 * h);
#pragma unroll
  for (int j = 0; j < 8; ++j) {
    int n = nB + j * 16 + ln;
    float bn = bias[n];
    v8bf sl = *(const v8bf*)(Bt + (size_t)n * KP + mw + 8 * h);
#pragma unroll
    for (int r = 0; r < 8; ++r) {
      float z = dv[r] * (acc[j][r] + (float)sl[r]) + bn;
      Xo[(size_t)(mw + 8 * h + r) * HDIM + n] = fmaxf(z, 0.f);
    }
  }
}

// ---------------- pooling: column-wise max & sum over 10000 rows -----------
__global__ __launch_bounds__(256) void pool_kernel(const float* __restrict__ X,
                                                   unsigned* __restrict__ gmax,
                                                   float* __restrict__ gsum) {
  const int t = threadIdx.x;                  // column 0..255
  const int rpb = (NNODES + gridDim.x - 1) / gridDim.x;
  const int r0 = blockIdx.x * rpb;
  const int r1 = (r0 + rpb < NNODES) ? r0 + rpb : NNODES;
  float mx = 0.f, sm = 0.f;                   // post-relu values are >= 0
  for (int m = r0; m < r1; ++m) {
    float v = X[(size_t)m * HDIM + t];
    mx = fmaxf(mx, v);
    sm += v;
  }
  atomicMax(gmax + t, __float_as_uint(mx));   // valid order for floats >= 0
  atomicAdd(gsum + t, sm);
}

// ---------------- MLP head: 512->256->128->3 -------------------------------
__global__ __launch_bounds__(256) void head_kernel(
    const unsigned* __restrict__ gmax, const float* __restrict__ gsum,
    const float* __restrict__ C1w, const float* __restrict__ c1b,
    const float* __restrict__ C2w, const float* __restrict__ c2b,
    const float* __restrict__ C3w, const float* __restrict__ c3b,
    float* __restrict__ out) {
  __shared__ float g[512];
  __shared__ float h1[256];
  __shared__ float h2[128];
  const int t = threadIdx.x;
  g[t]       = __uint_as_float(gmax[t]);
  g[256 + t] = gsum[t] * (1.0f / (float)NNODES);
  __syncthreads();
  float s = c1b[t];
  for (int i = 0; i < 512; ++i) s = fmaf(g[i], C1w[i * 256 + t], s);
  h1[t] = fmaxf(s, 0.f);
  __syncthreads();
  if (t < 128) {
    float s2 = c2b[t];
    for (int i = 0; i < 256; ++i) s2 = fmaf(h1[i], C2w[i * 128 + t], s2);
    h2[t] = fmaxf(s2, 0.f);
  }
  __syncthreads();
  if (t < 3) {
    float s3 = c3b[t];
    for (int i = 0; i < 128; ++i) s3 = fmaf(h2[i], C3w[i * 3 + t], s3);
    out[t] = s3;
  }
}

// ---------------------------------------------------------------------------
extern "C" void kernel_launch(void* const* d_in, const int* in_sizes, int n_in,
                              void* d_out, int out_size, void* d_ws, size_t ws_size,
                              hipStream_t stream) {
  const float* features = (const float*)d_in[0];
  const float* adj      = (const float*)d_in[1];
  const float* W0  = (const float*)d_in[2];
  const float* b0  = (const float*)d_in[3];
  const float* W1  = (const float*)d_in[4];
  const float* b1  = (const float*)d_in[5];
  const float* W2  = (const float*)d_in[6];
  const float* b2  = (const float*)d_in[7];
  const float* C1w = (const float*)d_in[8];
  const float* c1b = (const float*)d_in[9];
  const float* C2w = (const float*)d_in[10];
  const float* c2b = (const float*)d_in[11];
  const float* C3w = (const float*)d_in[12];
  const float* c3b = (const float*)d_in[13];

  // workspace carve-out (256B aligned slabs); adjb carved last so the base
  // layout is identical whether or not the cached-bf16 path is enabled.
  char* ws = (char*)d_ws;
  size_t off = 0;
  auto take = [&](size_t bytes) {
    size_t cur = off;
    off = (off + bytes + 255) & ~(size_t)255;
    return (void*)(ws + cur);
  };
  float*    dinv = (float*)take((size_t)KP * 4);
  __bf16*   Bt   = (__bf16*)take((size_t)HDIM * KP * 2);
  __bf16*   Wt0  = (__bf16*)take((size_t)HDIM * DIN * 2);
  __bf16*   Wt1  = (__bf16*)take((size_t)HDIM * HDIM * 2);
  __bf16*   Wt2  = (__bf16*)take((size_t)HDIM * HDIM * 2);
  float*    X1   = (float*)take((size_t)NNODES * HDIM * 4);
  float*    X2   = (float*)take((size_t)NNODES * HDIM * 4);
  unsigned* gmax = (unsigned*)take(HDIM * 4);
  float*    gsum = (float*)take(HDIM * 4);
  __bf16*   adjb = (__bf16*)take((size_t)NNODES * KP * 2);   // ~200MB, optional
  const bool cached = (ws_size >= off);   // deterministic across calls

  // prep (degree pass optionally fused with adj -> bf16 conversion)
  deg_cvt_kernel<<<NNODES, 256, 0, stream>>>(adj, dinv, cached ? adjb : (__bf16*)0);
  twt_kernel<<<(DIN * HDIM + 255) / 256, 256, 0, stream>>>(W0, Wt0, DIN, HDIM);
  twt_kernel<<<(HDIM * HDIM + 255) / 256, 256, 0, stream>>>(W1, Wt1, HDIM, HDIM);
  twt_kernel<<<(HDIM * HDIM + 255) / 256, 256, 0, stream>>>(W2, Wt2, HDIM, HDIM);
  init_kernel<<<16, 256, 0, stream>>>(Bt, gmax, gsum);

  const dim3 gg((NNODES + 127) / 128, 2);  // (79, 2)

  const float* Xin[3]   = {features, X1, X2};
  float*       Xout3[3] = {X1, X2, X1};
  const __bf16* Wts[3]  = {Wt0, Wt1, Wt2};
  const float*  bs[3]   = {b0, b1, b2};
  const int     Ks[3]   = {DIN, HDIM, HDIM};

  for (int l = 0; l < 3; ++l) {
    gemm_xw_kernel<<<gg, 256, 0, stream>>>(Xin[l], Wts[l], dinv, Bt, Ks[l]);
    if (cached)
      gemm_adj_kernel<true><<<gg, 256, 0, stream>>>(adjb, Bt, dinv, bs[l], Xout3[l]);
    else
      gemm_adj_kernel<false><<<gg, 256, 0, stream>>>(adj, Bt, dinv, bs[l], Xout3[l]);
  }

  // readout
  pool_kernel<<<80, 256, 0, stream>>>(X1, gmax, gsum);
  head_kernel<<<1, 256, 0, stream>>>(gmax, gsum, C1w, c1b, C2w, c2b, C3w, c3b,
                                     (float*)d_out);
}
